// AttentionSeq2SeqModel_59691455479922
// MI455X (gfx1250) — matrix operand order repaired and checked by reading, not compiled, not run on hardware
//
#include <hip/hip_runtime.h>

typedef __attribute__((ext_vector_type(16))) __bf16 v16bf;
typedef __attribute__((ext_vector_type(8)))  __bf16 v8bf;
typedef __attribute__((ext_vector_type(8)))  float  v8f;
typedef __attribute__((ext_vector_type(4)))  int    v4i;

constexpr int Bsz  = 2048;
constexpr int TENC = 96;
constexpr int Hdim = 512;
constexpr int TDEC = 48;
constexpr int ATTN = 14;
constexpr int GH   = 4 * Hdim;   // 2048 gate rows

static __device__ __forceinline__ unsigned short f32_to_bf16(float f) {
  union { float f; unsigned u; } v; v.f = f;
  unsigned u = v.u;
  u += 0x7fffu + ((u >> 16) & 1u);       // round-to-nearest-even
  return (unsigned short)(u >> 16);
}

static __device__ __forceinline__ v8f wmma_bf16(v16bf a, v16bf b, v8f c) {
  // D = A(16x32 bf16) x B(32x16 bf16) + C(16x16 f32)
  return __builtin_amdgcn_wmma_f32_16x16x32_bf16(false, a, false, b, (short)0, c,
                                                 false, false);
}

// --- CDNA5 async global->LDS copy (ASYNCcnt-tracked), with sync fallback ----
static __device__ __forceinline__ void async_copy_b128(const __bf16* gsrc,
                                                       __bf16* ldst) {
#if __has_builtin(__builtin_amdgcn_global_load_async_to_lds_b128)
  __builtin_amdgcn_global_load_async_to_lds_b128(
      (__attribute__((address_space(1))) v4i*)(void*)gsrc,
      (__attribute__((address_space(3))) v4i*)(void*)ldst, 0, 0);
#else
  *(v8bf*)ldst = *(const v8bf*)gsrc;
#endif
}

static __device__ __forceinline__ void wait_async0() {
#if __has_builtin(__builtin_amdgcn_s_wait_asynccnt)
  __builtin_amdgcn_s_wait_asynccnt(0);
#else
  asm volatile("s_wait_asynccnt 0x0" ::: "memory");
#endif
}

// ---------------------------------------------------------------------------
// One-time per call: weight f32->bf16 conversion, state init.
// ---------------------------------------------------------------------------
__global__ void init_kernel(const float* __restrict__ encWhh,
                            const float* __restrict__ decWhh,
                            const float* __restrict__ Wenc,
                            const float* __restrict__ Wdec,
                            const float* __restrict__ inputs,
                            unsigned short* __restrict__ encWhh16,
                            unsigned short* __restrict__ decWhh16,
                            unsigned short* __restrict__ Wenc16,
                            unsigned short* __restrict__ Wdec16,
                            unsigned short* __restrict__ h16init,
                            float* __restrict__ c,
                            float* __restrict__ totalAttn,
                            float* __restrict__ xdec) {
  const size_t stride = (size_t)gridDim.x * blockDim.x;
  const size_t g0 = (size_t)blockIdx.x * blockDim.x + threadIdx.x;
  for (size_t i = g0; i < (size_t)GH * Hdim; i += stride) {
    encWhh16[i] = f32_to_bf16(encWhh[i]);
    decWhh16[i] = f32_to_bf16(decWhh[i]);
  }
  for (size_t i = g0; i < (size_t)Hdim * Hdim; i += stride) {
    Wenc16[i] = f32_to_bf16(Wenc[i]);
    Wdec16[i] = f32_to_bf16(Wdec[i]);
  }
  for (size_t i = g0; i < (size_t)Bsz * Hdim; i += stride) {
    h16init[i] = 0;        // bf16 zero
    c[i] = 0.f;
  }
  for (size_t i = g0; i < (size_t)Bsz * ATTN; i += stride) totalAttn[i] = 0.f;
  for (size_t i = g0; i < (size_t)Bsz; i += stride)
    xdec[i] = inputs[i * TENC + (TENC - 1)];
}

// ---------------------------------------------------------------------------
// Fused LSTM step: gates = h_prev @ W_hh^T + x*W_ih^T + b, then c/h update.
// Block: 128 threads = 4 waves; wave w computes gate w for a 32(M)x64(N) tile.
// A-tiles double-buffered in LDS via async global->LDS copies; one barrier
// per K-chunk, copy of chunk k+1 overlaps WMMA of chunk k.
// ---------------------------------------------------------------------------
__global__ __launch_bounds__(128) void lstm_step_kernel(
    const unsigned short* __restrict__ hPrev16,   // B x H (bf16)
    const unsigned short* __restrict__ Whh16,     // 4H x H (bf16, row-major)
    const float* __restrict__ Wih,                // 4H
    const float* __restrict__ bias,               // 4H
    const float* __restrict__ x, int xStride,     // x value = x[b*xStride]
    float* __restrict__ c,                        // B x H (in-place)
    float* __restrict__ hNewF,                    // B x H (f32 out)
    unsigned short* __restrict__ hNew16)          // B x H (bf16 out)
{
  const int tid  = threadIdx.x;
  const int lane = tid & 31;
  const int wave = tid >> 5;                 // gate index 0..3
  const int m0   = blockIdx.x * 32;
  const int n0   = blockIdx.y * 64;

  __shared__ __bf16 ldsA[2][32 * 32];        // double-buffered A tile
  __shared__ float  ldsG[4][32][64];         // per-gate output tiles

  v8f acc[2][4] = {};

  // B operand: B[k,n] = W_hh[gate*H + n, k] -> row-major W_hh rows are B cols.
  const __bf16* Bg   = (const __bf16*)Whh16 + (size_t)(wave * Hdim) * Hdim;
  const int     koff = (lane >> 4) * 16;     // lanes 16-31 hold K=16..31
  const __bf16* brow[4];
#pragma unroll
  for (int ns = 0; ns < 4; ++ns)
    brow[ns] = Bg + (size_t)(n0 + ns * 16 + (lane & 15)) * Hdim + koff;

  const __bf16* Aglob = (const __bf16*)hPrev16;
  const int arow = lane & 15;
  const int ah   = (lane >> 4) * 16;         // A layout: lane<16 K 0..15, lane>=16 K 16..31
  const int ldr  = tid >> 2;                 // staging row 0..31
  const int ldc  = (tid & 3) * 8;            // staging col 0,8,16,24

  // prologue: stage K-chunk 0 into buffer 0
  async_copy_b128(Aglob + (size_t)(m0 + ldr) * Hdim + ldc, &ldsA[0][ldr * 32 + ldc]);

  constexpr int NCH = Hdim / 32;             // 16 K-chunks
  for (int q = 0; q < NCH; ++q) {
    wait_async0();
    __syncthreads();                         // chunk q resident in ldsA[q&1]
    if (q + 1 < NCH) {
      const int k1 = (q + 1) * 32;
      async_copy_b128(Aglob + (size_t)(m0 + ldr) * Hdim + k1 + ldc,
                      &ldsA[(q + 1) & 1][ldr * 32 + ldc]);
    }
    const __bf16* Ab = &ldsA[q & 1][0];
    const int k0 = q * 32;

    v8bf a0lo = *(const v8bf*)&Ab[arow * 32 + ah];
    v8bf a0hi = *(const v8bf*)&Ab[arow * 32 + 8 + ah];
    v8bf a1lo = *(const v8bf*)&Ab[(arow + 16) * 32 + ah];
    v8bf a1hi = *(const v8bf*)&Ab[(arow + 16) * 32 + 8 + ah];
    v16bf af0 = __builtin_shufflevector(a0lo, a0hi, 0,1,2,3,4,5,6,7,8,9,10,11,12,13,14,15);
    v16bf af1 = __builtin_shufflevector(a1lo, a1hi, 0,1,2,3,4,5,6,7,8,9,10,11,12,13,14,15);

    if (q + 1 < NCH) {                       // global_prefetch_b8 next B chunk
      __builtin_prefetch(brow[0] + k0 + 32, 0, 1);
      __builtin_prefetch(brow[2] + k0 + 32, 0, 1);
    }

#pragma unroll
    for (int ns = 0; ns < 4; ++ns) {
      v16bf bf = *(const v16bf*)(brow[ns] + k0);
      acc[0][ns] = wmma_bf16(af0, bf, acc[0][ns]);
      acc[1][ns] = wmma_bf16(af1, bf, acc[1][ns]);
    }
  }

  // C/D layout: lane<16 -> M=i, lane>=16 -> M=8+i; N = lane&15.
  {
    const int rb = (lane >> 4) * 8;
    const int cc = lane & 15;
#pragma unroll
    for (int ns = 0; ns < 4; ++ns)
#pragma unroll
      for (int i = 0; i < 8; ++i) {
        ldsG[wave][rb + i][ns * 16 + cc]      = acc[0][ns][i];
        ldsG[wave][16 + rb + i][ns * 16 + cc] = acc[1][ns][i];
      }
  }
  __syncthreads();

#pragma unroll
  for (int j = 0; j < 16; ++j) {
    const int e   = tid + j * 128;
    const int r   = e >> 6;
    const int col = e & 63;
    const int gb  = m0 + r;
    const int n   = n0 + col;
    const float xv = x[(size_t)gb * xStride];
    const float iz = ldsG[0][r][col] + bias[n]            + xv * Wih[n];
    const float fz = ldsG[1][r][col] + bias[Hdim + n]     + xv * Wih[Hdim + n];
    const float gz = ldsG[2][r][col] + bias[2 * Hdim + n] + xv * Wih[2 * Hdim + n];
    const float oz = ldsG[3][r][col] + bias[3 * Hdim + n] + xv * Wih[3 * Hdim + n];
    const float si = 1.f / (1.f + __expf(-iz));
    const float sf = 1.f / (1.f + __expf(-fz));
    const float so = 1.f / (1.f + __expf(-oz));
    const size_t idx = (size_t)gb * Hdim + n;
    const float cn = sf * c[idx] + si * tanhf(gz);
    c[idx] = cn;
    const float hn = so * tanhf(cn);
    hNewF[idx]  = hn;
    hNew16[idx] = f32_to_bf16(hn);
  }
}

// ---------------------------------------------------------------------------
// u = tanh(h_prev @ W_enc^T + h_new @ W_dec^T). Block tile 32(M) x 256(N),
// wave w owns columns [n0+64w, n0+64w+64). 32 K-chunks: chunks 0-15 use
// (h_prev, W_enc), 16-31 use (h_new, W_dec), same accumulators, same
// double-buffered async A pipeline.
// ---------------------------------------------------------------------------
__global__ __launch_bounds__(128) void attn_u_kernel(
    const unsigned short* __restrict__ hPrev16,
    const unsigned short* __restrict__ hNew16,
    const unsigned short* __restrict__ Wenc16,
    const unsigned short* __restrict__ Wdec16,
    float* __restrict__ tanhU)
{
  const int tid  = threadIdx.x;
  const int lane = tid & 31;
  const int wave = tid >> 5;
  const int m0   = blockIdx.x * 32;
  const int n0   = blockIdx.y * 256 + wave * 64;

  __shared__ __bf16 ldsA[2][32 * 32];

  v8f acc[2][4] = {};

  const int arow = lane & 15;
  const int ah   = (lane >> 4) * 16;
  const int ldr  = tid >> 2;
  const int ldc  = (tid & 3) * 8;
  const int koff = (lane >> 4) * 16;

  const __bf16* Aprev = (const __bf16*)hPrev16;
  const __bf16* Anew  = (const __bf16*)hNew16;
  const __bf16* WencB = (const __bf16*)Wenc16;
  const __bf16* WdecB = (const __bf16*)Wdec16;

  // prologue: stage chunk 0 (pass 0, k0 = 0)
  async_copy_b128(Aprev + (size_t)(m0 + ldr) * Hdim + ldc, &ldsA[0][ldr * 32 + ldc]);

  constexpr int NCH = 2 * (Hdim / 32);       // 32 chunks across both passes
  for (int q = 0; q < NCH; ++q) {
    wait_async0();
    __syncthreads();
    if (q + 1 < NCH) {
      const int qn = q + 1;
      const __bf16* An = (qn >= NCH / 2) ? Anew : Aprev;
      const int k1 = (qn & 15) * 32;
      async_copy_b128(An + (size_t)(m0 + ldr) * Hdim + k1 + ldc,
                      &ldsA[qn & 1][ldr * 32 + ldc]);
    }
    const __bf16* Ab = &ldsA[q & 1][0];
    const __bf16* Wm = (q >= NCH / 2) ? WdecB : WencB;
    const int k0 = (q & 15) * 32;

    v8bf a0lo = *(const v8bf*)&Ab[arow * 32 + ah];
    v8bf a0hi = *(const v8bf*)&Ab[arow * 32 + 8 + ah];
    v8bf a1lo = *(const v8bf*)&Ab[(arow + 16) * 32 + ah];
    v8bf a1hi = *(const v8bf*)&Ab[(arow + 16) * 32 + 8 + ah];
    v16bf af0 = __builtin_shufflevector(a0lo, a0hi, 0,1,2,3,4,5,6,7,8,9,10,11,12,13,14,15);
    v16bf af1 = __builtin_shufflevector(a1lo, a1hi, 0,1,2,3,4,5,6,7,8,9,10,11,12,13,14,15);

#pragma unroll
    for (int ns = 0; ns < 4; ++ns) {
      const __bf16* br =
          Wm + (size_t)(n0 + ns * 16 + (lane & 15)) * Hdim + koff + k0;
      v16bf bf = *(const v16bf*)br;
      acc[0][ns] = wmma_bf16(af0, bf, acc[0][ns]);
      acc[1][ns] = wmma_bf16(af1, bf, acc[1][ns]);
    }
  }

  const int rb = (lane >> 4) * 8;
  const int cc = lane & 15;
#pragma unroll
  for (int ns = 0; ns < 4; ++ns)
#pragma unroll
    for (int i = 0; i < 8; ++i) {
      tanhU[(size_t)(m0 + rb + i) * Hdim + n0 + ns * 16 + cc]      = tanhf(acc[0][ns][i]);
      tanhU[(size_t)(m0 + 16 + rb + i) * Hdim + n0 + ns * 16 + cc] = tanhf(acc[1][ns][i]);
    }
}

// ---------------------------------------------------------------------------
// scores = u @ W_val^T (14), softmax, total_attn +=, out = ctx @ W_fin^T + b.
// One wave (32 lanes) per batch row.
// ---------------------------------------------------------------------------
__global__ __launch_bounds__(256) void attn_fin_kernel(
    const float* __restrict__ tanhU,
    const float* __restrict__ hNewF,
    const float* __restrict__ Wval,     // ATT x H
    const float* __restrict__ Wfin,     // ATT + H (attn part first)
    const float* __restrict__ bfin,     // 1
    float* __restrict__ outputs,        // B x TDEC
    float* __restrict__ totalAttn,      // B x ATT
    float* __restrict__ xdec,           // B
    int t)
{
  const int gid  = blockIdx.x * blockDim.x + threadIdx.x;
  const int b    = gid >> 5;
  const int lane = gid & 31;
  if (b >= Bsz) return;

  const float* u = tanhU + (size_t)b * Hdim;
  float sc[ATTN];
#pragma unroll
  for (int a = 0; a < ATTN; ++a) {
    const float* wv = Wval + (size_t)a * Hdim;
    float p = 0.f;
    for (int k = lane; k < Hdim; k += 32) p += u[k] * wv[k];
    for (int off = 16; off > 0; off >>= 1) p += __shfl_xor(p, off, 32);
    sc[a] = p;
  }
  float m = sc[0];
#pragma unroll
  for (int a = 1; a < ATTN; ++a) m = fmaxf(m, sc[a]);
  float s = 0.f;
  float ee[ATTN];
#pragma unroll
  for (int a = 0; a < ATTN; ++a) { ee[a] = __expf(sc[a] - m); s += ee[a]; }
  const float inv = 1.f / s;

  if (lane < ATTN) totalAttn[(size_t)b * ATTN + lane] += ee[lane] * inv;

  float p = (lane < ATTN) ? ee[lane] * inv * Wfin[lane] : 0.f;
  const float* hrow = hNewF + (size_t)b * Hdim;
  for (int k = lane; k < Hdim; k += 32) p += hrow[k] * Wfin[ATTN + k];
  for (int off = 16; off > 0; off >>= 1) p += __shfl_xor(p, off, 32);
  if (lane == 0) {
    const float o = p + bfin[0];
    outputs[(size_t)b * TDEC + t] = o;
    xdec[b] = o;
  }
}

// ---------------------------------------------------------------------------
// Host side
// ---------------------------------------------------------------------------
extern "C" void kernel_launch(void* const* d_in, const int* in_sizes, int n_in,
                              void* d_out, int out_size, void* d_ws, size_t ws_size,
                              hipStream_t stream) {
  (void)in_sizes; (void)n_in; (void)out_size; (void)ws_size;
  const float* inputs = (const float*)d_in[0];
  const float* encWih = (const float*)d_in[2];
  const float* encWhh = (const float*)d_in[3];
  const float* encB   = (const float*)d_in[4];
  const float* decWih = (const float*)d_in[5];
  const float* decWhh = (const float*)d_in[6];
  const float* decB   = (const float*)d_in[7];
  const float* Wenc   = (const float*)d_in[8];
  const float* Wdec   = (const float*)d_in[9];
  const float* Wval   = (const float*)d_in[10];
  const float* Wfin   = (const float*)d_in[11];
  const float* bfin   = (const float*)d_in[12];

  float* outputs   = (float*)d_out;                       // B*TDEC
  float* totalAttn = (float*)d_out + (size_t)Bsz * TDEC;  // B*ATT

  char* ws = (char*)d_ws;
  size_t off = 0;
  auto take = [&](size_t bytes) { char* p = ws + off; off = (off + bytes + 255) & ~(size_t)255; return p; };
  unsigned short* encWhh16 = (unsigned short*)take((size_t)GH * Hdim * 2);
  unsigned short* decWhh16 = (unsigned short*)take((size_t)GH * Hdim * 2);
  unsigned short* Wenc16   = (unsigned short*)take((size_t)Hdim * Hdim * 2);
  unsigned short* Wdec16   = (unsigned short*)take((size_t)Hdim * Hdim * 2);
  unsigned short* h16[2];
  h16[0] = (unsigned short*)take((size_t)Bsz * Hdim * 2);
  h16[1] = (unsigned short*)take((size_t)Bsz * Hdim * 2);
  float* hF[2];
  hF[0] = (float*)take((size_t)Bsz * Hdim * 4);
  hF[1] = (float*)take((size_t)Bsz * Hdim * 4);
  float* cSt   = (float*)take((size_t)Bsz * Hdim * 4);
  float* tanhU = (float*)take((size_t)Bsz * Hdim * 4);
  float* xdec  = (float*)take((size_t)Bsz * 4);

  init_kernel<<<1024, 256, 0, stream>>>(encWhh, decWhh, Wenc, Wdec, inputs,
                                        encWhh16, decWhh16, Wenc16, Wdec16,
                                        h16[0], cSt, totalAttn, xdec);

  const dim3 gStep(Bsz / 32, Hdim / 64);    // 64 x 8
  const dim3 gU(Bsz / 32, Hdim / 256);      // 64 x 2
  int cur = 0;

  for (int t = 0; t < TENC; ++t) {
    lstm_step_kernel<<<gStep, 128, 0, stream>>>(
        h16[cur], encWhh16, encWih, encB,
        inputs + t, TENC,                    // x = inputs[b*TENC + t]
        cSt, hF[1 - cur], h16[1 - cur]);
    cur ^= 1;
  }

  for (int t = 0; t < TDEC; ++t) {
    lstm_step_kernel<<<gStep, 128, 0, stream>>>(
        h16[cur], decWhh16, decWih, decB,
        xdec, 1,                             // x = previous output
        cSt, hF[1 - cur], h16[1 - cur]);
    attn_u_kernel<<<gU, 128, 0, stream>>>(h16[cur], h16[1 - cur],
                                          Wenc16, Wdec16, tanhU);
    attn_fin_kernel<<<(Bsz * 32) / 256, 256, 0, stream>>>(
        tanhU, hF[1 - cur], Wval, Wfin, bfin, outputs, totalAttn, xdec, t);
    cur ^= 1;
  }
}
